// UpperTriangleEdgeAccuracyLoss_40355512714111
// MI455X (gfx1250) — compile-verified, matrix operand order checked by reading, not run on hardware
//
#include <hip/hip_runtime.h>

typedef __attribute__((ext_vector_type(16))) _Float16 v16h;
typedef __attribute__((ext_vector_type(8)))  float    v8f;

#define NSIDE 320
#define GSIZE (NSIDE * NSIDE)               // 102400
#define NBATCH 128
#define GROUPS_PER_IMG (GSIZE / 4)          // 25600 float4-groups per image
#define TOTAL_GROUPS (NBATCH * GROUPS_PER_IMG) // 3,276,800
#define EDGES_TOTAL 6533120.0               // 128 * 320*319/2

__global__ void UTEA_zero_ws(float* ws) { ws[0] = 0.0f; }

__global__ void __launch_bounds__(256)
UTEA_count_kernel(const float* __restrict__ inp,
                  const int*   __restrict__ tgt,
                  float* __restrict__ counter) {
    float p = 0.0f;  // per-thread match count (exact small integer)

    const unsigned stride = gridDim.x * blockDim.x;
    for (unsigned grp = blockIdx.x * blockDim.x + threadIdx.x;
         grp < TOTAL_GROUPS; grp += stride) {
        const unsigned b  = grp / GROUPS_PER_IMG;
        const unsigned g4 = (grp - b * GROUPS_PER_IMG) * 4u;

        const size_t base = (size_t)b * 3u * GSIZE + g4;
        // Coalesced 128-bit streams: 3 channels + target
        const float4 x0 = *(const float4*)(inp + base);
        const float4 x1 = *(const float4*)(inp + base + GSIZE);
        const float4 x2 = *(const float4*)(inp + base + 2 * (size_t)GSIZE);
        const int4   tv = *(const int4*)(tgt + (size_t)b * GSIZE + g4);

        // 320 % 4 == 0 -> a float4 group never crosses a matrix row
        const unsigned i  = g4 / NSIDE;
        const unsigned j0 = g4 - i * NSIDE;

        const float a0[4] = {x0.x, x0.y, x0.z, x0.w};
        const float a1[4] = {x1.x, x1.y, x1.z, x1.w};
        const float a2[4] = {x2.x, x2.y, x2.z, x2.w};
        const int   tt[4] = {tv.x, tv.y, tv.z, tv.w};

        #pragma unroll
        for (int u = 0; u < 4; ++u) {
            // argmax over C=3, first-max-wins (matches jnp.argmax)
            int   pred = 0;
            float best = a0[u];
            if (a1[u] > best) { best = a1[u]; pred = 1; }
            if (a2[u] > best) { best = a2[u]; pred = 2; }
            const bool valid = (j0 + u) > i;  // strict upper triangle
            p += (valid && (pred == tt[u])) ? 1.0f : 0.0f;
        }
    }

    // ---- wave32 reduction via WMMA ones-matrix trick ----
    // A = ones(16x32 f16); B[k,n] nonzero only in each lane's first K slot:
    //   B[0,n] = p[lane n] (lanes 0..15), B[8,n] = p[lane n+16] (lanes 16..31)
    // => D[m,n] = p[n] + p[n+16] for every m; every lane's d[0] = s[lane%16].
    v16h aones;
    #pragma unroll
    for (int k = 0; k < 16; ++k) aones[k] = (_Float16)1.0f;
    v16h bvec = {};
    bvec[0] = (_Float16)p;          // p <= 28 here: exact in f16
    v8f cz = {};
    v8f d = __builtin_amdgcn_wmma_f32_16x16x32_f16(
        /*neg_a=*/false, aones, /*neg_b=*/false, bvec,
        /*c_mod=*/(short)0, cz, /*reuse_a=*/false, /*reuse_b=*/false);

    float s = d[0];                 // pairwise-reduced (32 -> 16 values)
    s += __shfl_xor(s, 1);
    s += __shfl_xor(s, 2);
    s += __shfl_xor(s, 4);
    s += __shfl_xor(s, 8);          // full wave sum in every lane

    __shared__ float wsum[8];
    const unsigned lane = threadIdx.x & 31u;
    const unsigned wid  = threadIdx.x >> 5;
    if (lane == 0) wsum[wid] = s;
    __syncthreads();

    if (threadIdx.x == 0) {
        float tot = 0.0f;
        #pragma unroll
        for (int w = 0; w < 8; ++w) tot += wsum[w];
        atomicAdd(counter, tot);    // one atomic per block; integers < 2^24 -> exact
    }
}

__global__ void UTEA_finalize(const float* __restrict__ counter,
                              float* __restrict__ out) {
    out[0] = 1.0f - (float)((double)counter[0] / EDGES_TOTAL);
}

extern "C" void kernel_launch(void* const* d_in, const int* in_sizes, int n_in,
                              void* d_out, int out_size, void* d_ws, size_t ws_size,
                              hipStream_t stream) {
    const float* inp = (const float*)d_in[0];
    const int*   tgt = (const int*)d_in[1];
    float* out     = (float*)d_out;
    float* counter = (float*)d_ws;

    UTEA_zero_ws<<<1, 1, 0, stream>>>(counter);
    UTEA_count_kernel<<<2048, 256, 0, stream>>>(inp, tgt, counter);
    UTEA_finalize<<<1, 1, 0, stream>>>(counter, out);
}
